// GraphDecoder_40664750359188
// MI455X (gfx1250) — matrix operand order repaired and checked by reading, not compiled, not run on hardware
//
#include <hip/hip_runtime.h>

// ---------------- problem constants (match reference) ----------------
#define NN    11520     // B * NPG
#define NPGc  180
#define HFc   1024      // H * F_DIM
#define FDc   256       // F_DIM
#define K1PAD 448       // 436 padded to multiple of 64
#define MAXDEG 256

typedef __attribute__((ext_vector_type(16))) __bf16 v16bf;
typedef __attribute__((ext_vector_type(8)))  __bf16 v8bf;
typedef __attribute__((ext_vector_type(8)))  float  v8f;

__device__ __forceinline__ unsigned short f2bf(float f) {
    unsigned int u = __float_as_uint(f);
    return (unsigned short)((u + 0x7FFFu + ((u >> 16) & 1u)) >> 16);
}
__device__ __forceinline__ float bf2f(unsigned short h) {
    return __uint_as_float(((unsigned int)h) << 16);
}

// A fragment (16x32 bf16, ISA 7.12.2): lanes 0-15 hold row m, K={0..7,16..23};
// lanes 16-31 hold row m, K={8..15,24..31}.  aOff = half*8.
__device__ __forceinline__ v16bf ldA(const unsigned short* __restrict__ row,
                                     int k0, int aOff) {
    v8bf x0 = *reinterpret_cast<const v8bf*>(row + k0 + aOff);
    v8bf x1 = *reinterpret_cast<const v8bf*>(row + k0 + 16 + aOff);
    v16bf r;
#pragma unroll
    for (int i = 0; i < 8; ++i) { r[i] = x0[i]; r[i + 8] = x1[i]; }
    return r;
}
// B fragment (32x16): lane holds column n, K={bOff..bOff+7, 16+bOff..16+bOff+7}, bOff = half*8.
__device__ __forceinline__ v16bf ldB(const unsigned short* __restrict__ col,
                                     int k0, int bOff) {
    v8bf x0 = *reinterpret_cast<const v8bf*>(col + k0 + bOff);
    v8bf x1 = *reinterpret_cast<const v8bf*>(col + k0 + 16 + bOff);
    v16bf r;
#pragma unroll
    for (int i = 0; i < 8; ++i) { r[i] = x0[i]; r[i + 8] = x1[i]; }
    return r;
}
#define WMMA_BF16(a, b, c) \
    __builtin_amdgcn_wmma_f32_16x16x32_bf16(false, (a), false, (b), (short)0, (c), false, false)

// ---------------- WMMA GEMM: C[M,Nc] = A[M,K](bf16) * B[K,Nc](bf16, given as B^T) ----
// One wave per 32x32 C macro-tile: 2 A-frags x 2 B-frags -> 4 accumulators.
// Ping-pong pipeline, K % 64 == 0: stage A computes while stage B's loads fly
// and vice versa; buffer roles alternate by name so no register copies occur.
// EPI==0: store f32 into Cf.  EPI==1: bias+ReLU, store bf16 into Cb.
template <int EPI>
__global__ __launch_bounds__(32)
void k_wmma_gemm(const unsigned short* __restrict__ A,
                 const unsigned short* __restrict__ Bt,
                 const float* __restrict__ bias,
                 float* __restrict__ Cf,
                 unsigned short* __restrict__ Cb,
                 int K, int Nc) {
    const int lane = threadIdx.x;        // 0..31
    const int half = lane >> 4;          // 0 or 1
    const int l15  = lane & 15;
    const int mTile = blockIdx.y << 5;   // 32-row macro tile
    const int nTile = blockIdx.x << 5;   // 32-col macro tile
    const int aOff = half * 8;
    const int bOff = half * 8;

    const unsigned short* aRow0 = A  + (size_t)(mTile + l15) * K;
    const unsigned short* aRow1 = A  + (size_t)(mTile + 16 + l15) * K;
    const unsigned short* bCol0 = Bt + (size_t)(nTile + l15) * K;
    const unsigned short* bCol1 = Bt + (size_t)(nTile + 16 + l15) * K;

    v8f c00 = {}, c01 = {}, c10 = {}, c11 = {};

    // stage A holds k0, stage B holds k0+32
    v16bf a0A = ldA(aRow0, 0, aOff), a1A = ldA(aRow1, 0, aOff);
    v16bf b0A = ldB(bCol0, 0, bOff), b1A = ldB(bCol1, 0, bOff);
    v16bf a0B = ldA(aRow0, 32, aOff), a1B = ldA(aRow1, 32, aOff);
    v16bf b0B = ldB(bCol0, 32, bOff), b1B = ldB(bCol1, 32, bOff);

    for (int k0 = 0; k0 < K; k0 += 64) {
        // ---- stage A compute (stage B loads already in flight) ----
        c00 = WMMA_BF16(a0A, b0A, c00);
        c01 = WMMA_BF16(a0A, b1A, c01);
        c10 = WMMA_BF16(a1A, b0A, c10);
        c11 = WMMA_BF16(a1A, b1A, c11);
        if (k0 + 64 < K) {               // refill stage A; flies during stage B WMMAs
            a0A = ldA(aRow0, k0 + 64, aOff);
            a1A = ldA(aRow1, k0 + 64, aOff);
            b0A = ldB(bCol0, k0 + 64, bOff);
            b1A = ldB(bCol1, k0 + 64, bOff);
        }
        if (k0 + 256 < K) {              // stream toward near cache ~4 iters ahead
            __builtin_prefetch(aRow0 + k0 + 256, 0, 3);
            __builtin_prefetch(aRow1 + k0 + 256, 0, 3);
            __builtin_prefetch(bCol0 + k0 + 256, 0, 3);
            __builtin_prefetch(bCol1 + k0 + 256, 0, 3);
        }
        // ---- stage B compute ----
        c00 = WMMA_BF16(a0B, b0B, c00);
        c01 = WMMA_BF16(a0B, b1B, c01);
        c10 = WMMA_BF16(a1B, b0B, c10);
        c11 = WMMA_BF16(a1B, b1B, c11);
        if (k0 + 96 < K) {               // refill stage B; flies during next stage A
            a0B = ldA(aRow0, k0 + 96, aOff);
            a1B = ldA(aRow1, k0 + 96, aOff);
            b0B = ldB(bCol0, k0 + 96, bOff);
            b1B = ldB(bCol1, k0 + 96, bOff);
        }
    }

    // C/D layout: VGPR r, lanes 0-15 -> M=r, lanes 16-31 -> M=8+r; N = lane&15.
    const v8f* acc[2][2] = {{&c00, &c01}, {&c10, &c11}};
#pragma unroll
    for (int ti = 0; ti < 2; ++ti) {
#pragma unroll
        for (int tj = 0; tj < 2; ++tj) {
            const v8f& c = *acc[ti][tj];
            const int col = nTile + tj * 16 + l15;
#pragma unroll
            for (int r = 0; r < 8; ++r) {
                int row = mTile + ti * 16 + half * 8 + r;
                float v = c[r];
                if (EPI == 1) {
                    v += bias[col];
                    v = fmaxf(v, 0.f);
                    Cb[(size_t)row * Nc + col] = f2bf(v);
                } else {
                    Cf[(size_t)row * Nc + col] = v;
                }
            }
        }
    }
}

// ---------------- init GEMM: zc[64,256] = [z||cond] @ W_init + b_init ----------------
__global__ void k_zc(const float* __restrict__ z, const float* __restrict__ cond,
                     const float* __restrict__ W, const float* __restrict__ b,
                     float* __restrict__ zc) {
    int idx = blockIdx.x * 256 + threadIdx.x;   // 64*256 threads
    int bi = idx >> 8, c = idx & 255;
    const float* zp = z + bi * 256;
    const float* cp = cond + bi * 128;
    float acc = b[c];
    for (int k = 0; k < 256; ++k) acc = fmaf(zp[k], W[k * 256 + c], acc);
    for (int k = 0; k < 128; ++k) acc = fmaf(cp[k], W[(256 + k) * 256 + c], acc);
    zc[idx] = acc;
}

// ---------------- x0[N,448] bf16 = relu( zc[batch] || one-hot(order) ), pad 0 --------
__global__ void k_x0(const float* __restrict__ zc, unsigned short* __restrict__ X0) {
    long idx = (long)blockIdx.x * 256 + threadIdx.x;
    if (idx >= (long)NN * K1PAD) return;
    int c = (int)(idx % K1PAD);
    int n = (int)(idx / K1PAD);
    float v;
    if (c < 256)      v = fmaxf(zc[(n / NPGc) * 256 + c], 0.f);
    else if (c < 436) v = ((c - 256) == (n % NPGc)) ? 1.f : 0.f;
    else              v = 0.f;
    X0[idx] = f2bf(v);
}

// ---------------- weight transpose-convert: W[K,Nc] f32 -> Wt[Nc,Kpad] bf16 ----------
__global__ void k_wt(const float* __restrict__ W, unsigned short* __restrict__ Wt,
                     int K, int Nc, int Kpad) {
    long idx = (long)blockIdx.x * 256 + threadIdx.x;
    if (idx >= (long)Nc * Kpad) return;
    int k  = (int)(idx % Kpad);
    int nc = (int)(idx / Kpad);
    Wt[idx] = f2bf((k < K) ? W[(long)k * Nc + nc] : 0.f);
}

// ---------------- attention scores a_s/a_d [N,4] from Hf [N,1024] --------------------
__global__ void k_attn(const float* __restrict__ Hf, const float* __restrict__ ws,
                       const float* __restrict__ wd, float* __restrict__ as_o,
                       float* __restrict__ ad_o) {
    int idx = blockIdx.x * 256 + threadIdx.x;   // N*4
    if (idx >= NN * 4) return;
    int h = idx & 3, n = idx >> 2;
    const float* hp = Hf + (size_t)n * HFc + h * FDc;
    const float* sp = ws + h * FDc;
    const float* dp = wd + h * FDc;
    float s = 0.f, d = 0.f;
    for (int c = 0; c < FDc; ++c) { float v = hp[c]; s = fmaf(v, sp[c], s); d = fmaf(v, dp[c], d); }
    as_o[idx] = s; ad_o[idx] = d;
}

// ---------------- CSR build ----------------------------------------------------------
__global__ void k_zero(int* __restrict__ p, int n) {
    int i = blockIdx.x * 256 + threadIdx.x;
    if (i < n) p[i] = 0;
}
__global__ void k_count(const int* __restrict__ ei, int E, int etot, int* __restrict__ deg) {
    int e = blockIdx.x * 256 + threadIdx.x;
    if (e >= etot) return;
    int d = (e < E) ? ei[E + e] : (e - E);   // self loops appended
    atomicAdd(&deg[d], 1);
}
__global__ void k_scan_block(const int* __restrict__ deg, int* __restrict__ incl,
                             int* __restrict__ bsum) {
    __shared__ int sc[256];
    int t = threadIdx.x, g = blockIdx.x * 256 + t;
    sc[t] = deg[g];
    __syncthreads();
    for (int off = 1; off < 256; off <<= 1) {
        int x = (t >= off) ? sc[t - off] : 0;
        __syncthreads();
        sc[t] += x;
        __syncthreads();
    }
    incl[g] = sc[t];
    if (t == 255) bsum[blockIdx.x] = sc[255];
}
__global__ void k_scan_sums(const int* __restrict__ bsum, int* __restrict__ boff,
                            int* __restrict__ rowptr, int nb, int etot) {
    if (blockIdx.x == 0 && threadIdx.x == 0) {
        int run = 0;
        for (int b = 0; b < nb; ++b) { boff[b] = run; run += bsum[b]; }
        rowptr[NN] = etot;
    }
}
__global__ void k_scan_fix(const int* __restrict__ incl, const int* __restrict__ deg,
                           const int* __restrict__ boff, int* __restrict__ rowptr) {
    int g = blockIdx.x * 256 + threadIdx.x;
    rowptr[g] = incl[g] - deg[g] + boff[blockIdx.x];
}
__global__ void k_scatter(const int* __restrict__ ei, int E, int etot,
                          const int* __restrict__ rowptr, int* __restrict__ cursor,
                          int* __restrict__ edgeSrc) {
    int e = blockIdx.x * 256 + threadIdx.x;
    if (e >= etot) return;
    int s, d;
    if (e < E) { s = ei[e]; d = ei[E + e]; } else { s = e - E; d = e - E; }
    int pos = atomicAdd(&cursor[d], 1);
    edgeSrc[rowptr[d] + pos] = s;
}

// ---------------- per-dst softmax + aggregation; bias+ReLU+bf16 epilogue -------------
__global__ __launch_bounds__(128)
void k_aggregate(const float* __restrict__ Hf,
                 const float* __restrict__ as_o, const float* __restrict__ ad_o,
                 const int* __restrict__ rowptr, const int* __restrict__ edgeSrc,
                 const float* __restrict__ bias, unsigned short* __restrict__ Xout) {
    __shared__ float shp[4][MAXDEG];
    __shared__ float red[128];
    __shared__ float sMax[4], sRcp[4];
    const int d = blockIdx.x;
    const int t = threadIdx.x;
    const int beg = rowptr[d];
    int deg = rowptr[d + 1] - beg;
    if (deg > MAXDEG) deg = MAXDEG;

    float adv[4];
#pragma unroll
    for (int h = 0; h < 4; ++h) adv[h] = ad_o[d * 4 + h];

    // pass 1: e = LeakyReLU(a_s[src] + a_d[dst]); track max per head
    float mx[4] = {-1e30f, -1e30f, -1e30f, -1e30f};
    for (int i = t; i < deg; i += 128) {
        int s = edgeSrc[beg + i];
#pragma unroll
        for (int h = 0; h < 4; ++h) {
            float e = as_o[s * 4 + h] + adv[h];
            e = (e > 0.f) ? e : 0.2f * e;
            shp[h][i] = e;
            mx[h] = fmaxf(mx[h], e);
        }
    }
    for (int h = 0; h < 4; ++h) {
        red[t] = mx[h]; __syncthreads();
        for (int s2 = 64; s2 > 0; s2 >>= 1) {
            if (t < s2) red[t] = fmaxf(red[t], red[t + s2]);
            __syncthreads();
        }
        if (t == 0) sMax[h] = red[0];
        __syncthreads();
    }
    // pass 2: p = exp(e - max); sum per head
    float sm[4] = {0.f, 0.f, 0.f, 0.f};
    for (int i = t; i < deg; i += 128) {
#pragma unroll
        for (int h = 0; h < 4; ++h) {
            float pv = __expf(shp[h][i] - sMax[h]);
            shp[h][i] = pv;
            sm[h] += pv;
        }
    }
    for (int h = 0; h < 4; ++h) {
        red[t] = sm[h]; __syncthreads();
        for (int s2 = 64; s2 > 0; s2 >>= 1) {
            if (t < s2) red[t] += red[t + s2];
            __syncthreads();
        }
        if (t == 0) sRcp[h] = 1.f / (red[0] + 1e-16f);
        __syncthreads();
    }
    for (int i = t; i < deg; i += 128) {
#pragma unroll
        for (int h = 0; h < 4; ++h) shp[h][i] *= sRcp[h];
    }
    __syncthreads();

    // pass 3: thread t owns features [t*8, t*8+8) (all within one head: 8 | 256)
    const int f0 = t * 8;
    const int h0 = f0 >> 8;
    float acc[8] = {0.f, 0.f, 0.f, 0.f, 0.f, 0.f, 0.f, 0.f};
    for (int i = 0; i < deg; ++i) {
        int s = edgeSrc[beg + i];
        float al = shp[h0][i];
        const float* hp = Hf + (size_t)s * HFc + f0;
#pragma unroll
        for (int j = 0; j < 8; ++j) acc[j] = fmaf(al, hp[j], acc[j]);
    }
    unsigned short* xo = Xout + (size_t)d * HFc + f0;
#pragma unroll
    for (int j = 0; j < 8; ++j) {
        float v = acc[j] + bias[f0 + j];
        xo[j] = f2bf(fmaxf(v, 0.f));
    }
}

// ---------------- tiny head final: out[N,O] = G[N,256](bf16) @ W[256,O] + b ----------
__global__ void k_head_final(const unsigned short* __restrict__ G,
                             const float* __restrict__ W, const float* __restrict__ b,
                             float* __restrict__ out, int O) {
    int idx = blockIdx.x * 256 + threadIdx.x;   // N*O
    if (idx >= NN * O) return;
    int o = idx % O, n = idx / O;
    const unsigned short* gp = G + (size_t)n * FDc;
    float acc = b[o];
    for (int c = 0; c < FDc; ++c) acc = fmaf(bf2f(gp[c]), W[c * O + o], acc);
    out[idx] = acc;
}

// =====================================================================================
extern "C" void kernel_launch(void* const* d_in, const int* in_sizes, int n_in,
                              void* d_out, int out_size, void* d_ws, size_t ws_size,
                              hipStream_t stream) {
    const float* z      = (const float*)d_in[0];
    const float* cond   = (const float*)d_in[1];
    const int*   ei     = (const int*)  d_in[2];
    // d_in[3] = batch (order derivable: n % 180, segments sorted & equal-sized)
    const float* W_init = (const float*)d_in[4];
    const float* b_init = (const float*)d_in[5];
    const float* W1  = (const float*)d_in[6];
    const float* as1 = (const float*)d_in[7];
    const float* ad1 = (const float*)d_in[8];
    const float* b1  = (const float*)d_in[9];
    const float* W2  = (const float*)d_in[10];
    const float* as2 = (const float*)d_in[11];
    const float* ad2 = (const float*)d_in[12];
    const float* b2  = (const float*)d_in[13];
    const float* W3  = (const float*)d_in[14];
    const float* as3 = (const float*)d_in[15];
    const float* ad3 = (const float*)d_in[16];
    const float* b3  = (const float*)d_in[17];
    const float* Wp  = (const float*)d_in[18];
    const float* bp  = (const float*)d_in[19];
    const float* Wfp = (const float*)d_in[20];
    const float* bfp = (const float*)d_in[21];
    const float* Ws  = (const float*)d_in[22];
    const float* bs  = (const float*)d_in[23];
    const float* Wfs = (const float*)d_in[24];
    const float* bfs = (const float*)d_in[25];
    const float* Wt  = (const float*)d_in[26];
    const float* bt  = (const float*)d_in[27];
    const float* Wft = (const float*)d_in[28];
    const float* bft = (const float*)d_in[29];
    float* out = (float*)d_out;

    const int E    = in_sizes[2] / 2;
    const int Etot = E + NN;

    // ---- workspace carve (256B aligned) ----
    char* p = (char*)d_ws;
    auto carve = [&](size_t bytes) -> char* {
        char* r = p;
        p += (bytes + 255) & ~(size_t)255;
        return r;
    };
    float*          zc   = (float*)         carve((size_t)64 * 256 * 4);
    unsigned short* X0   = (unsigned short*)carve((size_t)NN * K1PAD * 2);
    unsigned short* X1   = (unsigned short*)carve((size_t)NN * HFc * 2);
    float*          Hf   = (float*)         carve((size_t)NN * HFc * 4);
    unsigned short* W1t  = (unsigned short*)carve((size_t)HFc * K1PAD * 2);
    unsigned short* W2t  = (unsigned short*)carve((size_t)HFc * HFc * 2);
    unsigned short* W3t  = (unsigned short*)carve((size_t)HFc * HFc * 2);
    unsigned short* Wpt  = (unsigned short*)carve((size_t)FDc * HFc * 2);
    unsigned short* Wst  = (unsigned short*)carve((size_t)FDc * HFc * 2);
    unsigned short* Wtt  = (unsigned short*)carve((size_t)FDc * HFc * 2);
    float*          asb  = (float*)         carve((size_t)NN * 4 * 4);
    float*          adb  = (float*)         carve((size_t)NN * 4 * 4);
    int*            deg  = (int*)           carve((size_t)NN * 4);
    int*            incl = (int*)           carve((size_t)NN * 4);
    int*            bsum = (int*)           carve(64 * 4);
    int*            boff = (int*)           carve(64 * 4);
    int*            rowp = (int*)           carve((size_t)(NN + 1) * 4);
    int*            curs = (int*)           carve((size_t)NN * 4);
    int*            eSrc = (int*)           carve((size_t)Etot * 4);
    unsigned short* G    = (unsigned short*)Hf;   // head temp reuses Hf space

    const int NB_N   = NN / 256;                  // 45 (exact)
    const int NB_E   = (Etot + 255) / 256;
    const dim3 gGemmBig(HFc / 32, NN / 32);       // (32, 360)
    const dim3 gGemmHead(FDc / 32, NN / 32);      // (8, 360)

    // ---- init: zc, x0, weight converts ----
    k_zc<<<64, 256, 0, stream>>>(z, cond, W_init, b_init, zc);
    {
        long tot = (long)NN * K1PAD;
        k_x0<<<(unsigned)((tot + 255) / 256), 256, 0, stream>>>(zc, X0);
    }
    auto wt = [&](const float* W, unsigned short* Wt_, int K, int Nc, int Kpad) {
        long tot = (long)Nc * Kpad;
        k_wt<<<(unsigned)((tot + 255) / 256), 256, 0, stream>>>(W, Wt_, K, Nc, Kpad);
    };
    wt(W1, W1t, 436, HFc, K1PAD);
    wt(W2, W2t, HFc, HFc, HFc);
    wt(W3, W3t, HFc, HFc, HFc);
    wt(Wp, Wpt, HFc, FDc, HFc);
    wt(Ws, Wst, HFc, FDc, HFc);
    wt(Wt, Wtt, HFc, FDc, HFc);

    // ---- CSR by destination (edges constant across layers) ----
    k_zero<<<NB_N, 256, 0, stream>>>(deg, NN);
    k_zero<<<NB_N, 256, 0, stream>>>(curs, NN);
    k_count<<<NB_E, 256, 0, stream>>>(ei, E, Etot, deg);
    k_scan_block<<<NB_N, 256, 0, stream>>>(deg, incl, bsum);
    k_scan_sums<<<1, 64, 0, stream>>>(bsum, boff, rowp, NB_N, Etot);
    k_scan_fix<<<NB_N, 256, 0, stream>>>(incl, deg, boff, rowp);
    k_scatter<<<NB_E, 256, 0, stream>>>(ei, E, Etot, rowp, curs, eSrc);

    // ---- GAT layer 1 ----
    k_wmma_gemm<0><<<gGemmBig, 32, 0, stream>>>(X0, W1t, nullptr, Hf, nullptr, K1PAD, HFc);
    k_attn<<<(NN * 4) / 256, 256, 0, stream>>>(Hf, as1, ad1, asb, adb);
    k_aggregate<<<NN, 128, 0, stream>>>(Hf, asb, adb, rowp, eSrc, b1, X1);

    // ---- GAT layer 2 ----
    k_wmma_gemm<0><<<gGemmBig, 32, 0, stream>>>(X1, W2t, nullptr, Hf, nullptr, HFc, HFc);
    k_attn<<<(NN * 4) / 256, 256, 0, stream>>>(Hf, as2, ad2, asb, adb);
    k_aggregate<<<NN, 128, 0, stream>>>(Hf, asb, adb, rowp, eSrc, b2, X1);

    // ---- GAT layer 3 ----
    k_wmma_gemm<0><<<gGemmBig, 32, 0, stream>>>(X1, W3t, nullptr, Hf, nullptr, HFc, HFc);
    k_attn<<<(NN * 4) / 256, 256, 0, stream>>>(Hf, as3, ad3, asb, adb);
    k_aggregate<<<NN, 128, 0, stream>>>(Hf, asb, adb, rowp, eSrc, b3, X1);

    // ---- output heads (sequential; G buffer reused) ----
    // pos: [N,2]
    k_wmma_gemm<1><<<gGemmHead, 32, 0, stream>>>(X1, Wpt, bp, nullptr, G, HFc, FDc);
    k_head_final<<<(NN * 2 + 255) / 256, 256, 0, stream>>>(G, Wfp, bfp, out, 2);
    // size: [N,2]
    k_wmma_gemm<1><<<gGemmHead, 32, 0, stream>>>(X1, Wst, bs, nullptr, G, HFc, FDc);
    k_head_final<<<(NN * 2 + 255) / 256, 256, 0, stream>>>(G, Wfs, bfs, out + (size_t)NN * 2, 2);
    // theta: [N,1]
    k_wmma_gemm<1><<<gGemmHead, 32, 0, stream>>>(X1, Wtt, bt, nullptr, G, HFc, FDc);
    k_head_final<<<(NN * 1 + 255) / 256, 256, 0, stream>>>(G, Wft, bft, out + (size_t)NN * 4, 1);

    (void)n_in; (void)out_size; (void)ws_size;
}
